// PGN_66477503807744
// MI455X (gfx1250) — matrix operand order repaired and checked by reading, not compiled, not run on hardware
//
#include <hip/hip_runtime.h>
#include <stdint.h>

// ---------------------------------------------------------------------------
// Pointer-generator decoder, CDNA5 (gfx1250, wave32).
// bf16 WMMA (v_wmma_f32_16x16x32_bf16, f32 accumulate) for all GEMMs;
// B operands pre-packed once into the wave32 WMMA VGPR layout; the dominant
// logits GEMM processes 4 N-tiles per wave to amortize A loads (A traffic
// 51.2 -> 12.8 MB/step; B = 51 MB/step of bf16 Wfc, L2-resident in 192 MB L2).
// Fused per-step kernels: 10 launches/step on the 100-step serial chain.
// Requires ws_size >= ~80 MB.
// ---------------------------------------------------------------------------

typedef __attribute__((ext_vector_type(16))) __bf16 v16bf;
typedef __attribute__((ext_vector_type(8)))  float  v8f;

#define BB   16
#define SS   400
#define TT   100
#define DD   512
#define EE   256
#define VV   50000
#define OOVN 50
#define EXTV (VV + OOVN)
#define G3D  1536          // 3*D
#define DE   768           // D+E
#define TILE_WORDS 256     // 32 lanes * 8 dwords per (tile, 16x32 k-block)

#define HD __device__ __forceinline__

HD unsigned short f2bf(float f) {
    union { float f; uint32_t u; } c; c.f = f;
    uint32_t r = c.u + 0x7FFFu + ((c.u >> 16) & 1u);   // round-to-nearest-even
    return (unsigned short)(r >> 16);
}
HD uint32_t pack2bf(float a, float b) {
    return (uint32_t)f2bf(a) | ((uint32_t)f2bf(b) << 16);
}
HD v16bf load16bf(const uint32_t* p) {
    union { uint4 u[2]; v16bf v; } t;
    t.u[0] = *(const uint4*)(p);
    t.u[1] = *(const uint4*)(p + 4);
    return t.v;
}
HD float sigmoidf(float x) { return 1.0f / (1.0f + __expf(-x)); }

// ---- block reductions (blockDim power of two) -----------------------------
HD float block_reduce_sum(float v, float* red) {
    int tid = threadIdx.x;
    red[tid] = v; __syncthreads();
    for (int off = blockDim.x >> 1; off > 0; off >>= 1) {
        if (tid < off) red[tid] += red[tid + off];
        __syncthreads();
    }
    float r = red[0]; __syncthreads();
    return r;
}
HD float block_reduce_max(float v, float* red) {
    int tid = threadIdx.x;
    red[tid] = v; __syncthreads();
    for (int off = blockDim.x >> 1; off > 0; off >>= 1) {
        if (tid < off) red[tid] = fmaxf(red[tid], red[tid + off]);
        __syncthreads();
    }
    float r = red[0]; __syncthreads();
    return r;
}

// ---------------------------------------------------------------------------
// WMMA operand layout helpers (from CDNA5 ISA, wave32):
// A (16-bit 16xK): lane l -> row m=l&15, khalf=l>>4; dword i -> K pair
//   kb*32 + 2i + (i>=4?8:0) + 8*khalf.
// B (16-bit Kx16): lane l -> col n=l&15, khalf=l>>4; dword i -> K pair
//   kb*32 + 2i + (i>=4?8:0) + 8*khalf.
// Word index: ((tile*KB + kb)*32 + lane)*8 + i
// ---------------------------------------------------------------------------
HD int a_k_of(int kb, int l, int i) { return kb * 32 + 2 * i + ((i >= 4) ? 8 : 0) + 8 * (l >> 4); }
HD int b_k_of(int kb, int l, int i) { return kb * 32 + 2 * i + ((i >= 4) ? 8 : 0) + 8 * (l >> 4); }

__global__ void k_pack_a(const float* __restrict__ src, uint32_t* __restrict__ dst,
                         int MT, int KB, int K) {
    size_t idx = (size_t)blockIdx.x * blockDim.x + threadIdx.x;
    size_t total = (size_t)MT * KB * TILE_WORDS;
    if (idx >= total) return;
    int i  = idx & 7;
    int l  = (idx >> 3) & 31;
    size_t t2 = idx >> 8;
    int kb = (int)(t2 % KB);
    int mt = (int)(t2 / KB);
    int m  = mt * 16 + (l & 15);
    int k  = a_k_of(kb, l, i);
    const float* row = src + (size_t)m * K;
    dst[idx] = pack2bf(row[k], row[k + 1]);
}

// NTpad tiles emitted; columns >= Nreal zero-filled (lets GEMM run 4 tiles/wave).
__global__ void k_pack_b(const float* __restrict__ src, uint32_t* __restrict__ dst,
                         int NTpad, int KB, int N, int Nreal) {
    size_t idx = (size_t)blockIdx.x * blockDim.x + threadIdx.x;
    size_t total = (size_t)NTpad * KB * TILE_WORDS;
    if (idx >= total) return;
    int i  = idx & 7;
    int l  = (idx >> 3) & 31;
    size_t t2 = idx >> 8;
    int kb = (int)(t2 % KB);
    int nt = (int)(t2 / KB);
    int n  = nt * 16 + (l & 15);
    if (n >= Nreal) { dst[idx] = 0u; return; }
    int k  = b_k_of(kb, l, i);
    dst[idx] = pack2bf(src[(size_t)k * N + n], src[(size_t)(k + 1) * N + n]);
}

// ---------------------------------------------------------------------------
// WMMA GEMM, 4 N-tiles per wave: one A load feeds 4 wmma per k-block.
// blockDim = 128 (4 waves). grid.x covers tile-groups, grid.y covers M tiles.
// NTpad must be a multiple of 4 (B buffer padded/zero-filled accordingly).
// ---------------------------------------------------------------------------
__global__ void k_gemm_bf16x4(const uint32_t* __restrict__ Ap,
                              const uint32_t* __restrict__ Bp,
                              const float* __restrict__ bias,
                              float* __restrict__ out,
                              int NTreal, int NTpad, int KB, int ldN) {
    int lane = threadIdx.x & 31;
    int wave = threadIdx.x >> 5;
    int grp  = blockIdx.x * (blockDim.x >> 5) + wave;   // group of 4 tiles
    int mt   = blockIdx.y;
    int nt0  = grp * 4;
    if (nt0 >= NTpad) return;                           // wave-uniform
    const size_t tileStride = (size_t)KB * TILE_WORDS;
    const uint32_t* ap = Ap + ((size_t)mt * KB * 32 + lane) * 8;
    const uint32_t* bp = Bp + (size_t)nt0 * tileStride + (size_t)lane * 8;
    v8f acc0 = {}, acc1 = {}, acc2 = {}, acc3 = {};
    for (int kb = 0; kb < KB; ++kb) {
        v16bf a  = load16bf(ap); ap += TILE_WORDS;
        v16bf b0 = load16bf(bp);
        v16bf b1 = load16bf(bp + tileStride);
        v16bf b2 = load16bf(bp + 2 * tileStride);
        v16bf b3 = load16bf(bp + 3 * tileStride);
        bp += TILE_WORDS;
        acc0 = __builtin_amdgcn_wmma_f32_16x16x32_bf16(false, a, false, b0, (short)0, acc0, false, false);
        acc1 = __builtin_amdgcn_wmma_f32_16x16x32_bf16(false, a, false, b1, (short)0, acc1, false, false);
        acc2 = __builtin_amdgcn_wmma_f32_16x16x32_bf16(false, a, false, b2, (short)0, acc2, false, false);
        acc3 = __builtin_amdgcn_wmma_f32_16x16x32_bf16(false, a, false, b3, (short)0, acc3, false, false);
    }
    int col = lane & 15;
    int mb  = (lane >> 4) * 8;               // C/D: lanes16-31 hold M=r+8
    v8f accs[4] = {acc0, acc1, acc2, acc3};
#pragma unroll
    for (int j = 0; j < 4; ++j) {
        int nt = nt0 + j;
        if (nt < NTreal) {                   // wave-uniform guard
            int   n  = nt * 16 + col;
            float bv = bias ? bias[n] : 0.0f;
            float* o = out + (size_t)(mt * 16 + mb) * ldN + n;
#pragma unroll
            for (int r = 0; r < 8; ++r) o[(size_t)r * ldN] = accs[j][r] + bv;
        }
    }
}

// ---------------------------------------------------------------------------
// Fused small kernels
// ---------------------------------------------------------------------------
__global__ void k_copy_f32(const float* __restrict__ src, float* __restrict__ dst, int n) {
    int i = blockIdx.x * blockDim.x + threadIdx.x;
    if (i < n) dst[i] = src[i];
}

// Pack gin = [ctx_prev, emb[x_t]] directly into A layout (no f32 gin buffer).
HD float gin_val(const float* ctx, const float* emb, const int* dec_inp, int t, int m, int k) {
    if (k < DD) return ctx[m * DD + k];
    int row = dec_inp[m * TT + t];
    return emb[(size_t)row * EE + (k - DD)];
}
__global__ void k_pack_gin(const float* __restrict__ ctx, const float* __restrict__ emb,
                           const int* __restrict__ dec_inp, int t,
                           uint32_t* __restrict__ ginAP) {
    int idx = blockIdx.x * blockDim.x + threadIdx.x;     // (DE/32)*256 = 6144
    if (idx >= (DE / 32) * TILE_WORDS) return;
    int i  = idx & 7;
    int l  = (idx >> 3) & 31;
    int kb = idx >> 8;
    int m  = l & 15;
    int k  = a_k_of(kb, l, i);
    ginAP[idx] = pack2bf(gin_val(ctx, emb, dec_inp, t, m, k),
                         gin_val(ctx, emb, dec_inp, t, m, k + 1));
}

// GRU gates (Keras reset_after=True, [z,r,h]) fused with bf16 A-repack of h.
// One thread per packed word = one (m,k),(m,k+1) pair -> race-free in-place h.
HD float gru_one(const float* xg, const float* hg, const float* h, int b, int d) {
    const float* xr = xg + (size_t)b * G3D;
    const float* hr = hg + (size_t)b * G3D;
    float z  = sigmoidf(xr[d] + hr[d]);
    float r  = sigmoidf(xr[DD + d] + hr[DD + d]);
    float hh = tanhf(xr[2 * DD + d] + r * hr[2 * DD + d]);
    float ho = h[(size_t)b * DD + d];
    return z * ho + (1.0f - z) * hh;
}
__global__ void k_gru_gates_pack(const float* __restrict__ xg, const float* __restrict__ hg,
                                 float* __restrict__ h, uint32_t* __restrict__ hAP) {
    int idx = blockIdx.x * blockDim.x + threadIdx.x;     // (DD/32)*256 = 4096
    if (idx >= (DD / 32) * TILE_WORDS) return;
    int i  = idx & 7;
    int l  = (idx >> 3) & 31;
    int kb = idx >> 8;
    int m  = l & 15;
    int k  = a_k_of(kb, l, i);
    float h0 = gru_one(xg, hg, h, m, k);
    float h1 = gru_one(xg, hg, h, m, k + 1);
    h[(size_t)m * DD + k]     = h0;
    h[(size_t)m * DD + k + 1] = h1;
    hAP[idx] = pack2bf(h0, h1);
}

// Per-row max & sum-of-exp over V logits. 16 blocks x 1024 threads.
__global__ void k_softmax_stats(const float* __restrict__ logits,
                                float* __restrict__ rowmax, float* __restrict__ rowsum) {
    __shared__ float red[1024];
    int b = blockIdx.x;
    const float* row = logits + (size_t)b * VV;
    float m = -3.4e38f;
    for (int v = threadIdx.x; v < VV; v += blockDim.x) m = fmaxf(m, row[v]);
    float M = block_reduce_max(m, red);
    float s = 0.0f;
    for (int v = threadIdx.x; v < VV; v += blockDim.x) s += __expf(row[v] - M);
    float S = block_reduce_sum(s, red);
    if (threadIdx.x == 0) { rowmax[b] = M; rowsum[b] = S; }
}

// Fused attention: scores (Va.tanh(enc_feat+hW2)+bV) -> softmax over S ->
// ctx_new -> p_gen.  One block (512 thr = 16 waves) per batch row.
__global__ void k_attention(const float* __restrict__ enc_feat,
                            const float* __restrict__ hW2,
                            const float* __restrict__ enc_output,
                            const float* __restrict__ Va, const float* __restrict__ bV,
                            const float* __restrict__ h,
                            const float* __restrict__ ctx_prev,
                            const float* __restrict__ emb,
                            const int* __restrict__ dec_inp, int t,
                            const float* __restrict__ wc, const float* __restrict__ wsv,
                            const float* __restrict__ wx, const float* __restrict__ bp_,
                            float* __restrict__ attn, float* __restrict__ ctx_new,
                            float* __restrict__ pgen, int do_pgen) {
    __shared__ float red[512];
    __shared__ float sa[SS];
    int b    = blockIdx.x;
    int tid  = threadIdx.x;
    int lane = tid & 31;
    int wv   = tid >> 5;                       // 16 waves (wave32)
    const float* hw = hW2 + (size_t)b * DD;
    // ---- scores: each wave owns s = wv, wv+16, ... ; 512-dot across 32 lanes
    for (int s = wv; s < SS; s += 16) {
        const float* ef = enc_feat + ((size_t)b * SS + s) * DD;
        float acc = 0.0f;
        for (int a = lane; a < DD; a += 32)
            acc += tanhf(ef[a] + hw[a]) * Va[a];
        for (int off = 16; off > 0; off >>= 1)
            acc += __shfl_down(acc, off, 32);
        if (lane == 0) sa[s] = acc + bV[0];
    }
    __syncthreads();
    // ---- softmax over S
    float m = (tid < SS) ? sa[tid] : -3.4e38f;
    float M = block_reduce_max(m, red);
    float e = 0.0f;
    if (tid < SS) { e = __expf(sa[tid] - M); }
    float S = block_reduce_sum(e, red);
    float inv = 1.0f / S;
    if (tid < SS) {
        float a = e * inv;
        sa[tid] = a;
        attn[(size_t)b * SS + tid] = a;
    }
    __syncthreads();
    // ---- ctx_new[b,d] = sum_s attn * enc_output
    float ctxd = 0.0f;
    {
        int d = tid;                           // 512 threads == DD
        const float* eo = enc_output + (size_t)b * SS * DD + d;
        for (int s = 0; s < SS; ++s) ctxd += sa[s] * eo[(size_t)s * DD];
        ctx_new[(size_t)b * DD + d] = ctxd;
    }
    // ---- p_gen = sigmoid(ctx_new.wc + h.ws + [ctx_prev,emb_row].wx + bp)
    if (do_pgen) {
        int d = tid;
        float p = ctxd * wc[d] + h[(size_t)b * DD + d] * wsv[d]
                + ctx_prev[(size_t)b * DD + d] * wx[d];
        if (d < EE) {
            int row = dec_inp[b * TT + t];
            p += emb[(size_t)row * EE + d] * wx[DD + d];
        }
        float tot = block_reduce_sum(p, red);
        if (tid == 0) pgen[b] = sigmoidf(tot + bp_[0]);
    }
}

// out[b, t, v] = pgen*softmax(logits) for v<V, else 0 (OOV slots).
__global__ void k_write_out(const float* __restrict__ logits,
                            const float* __restrict__ rowmax, const float* __restrict__ rowsum,
                            const float* __restrict__ pgen,
                            float* __restrict__ out, int t) {
    size_t idx = (size_t)blockIdx.x * blockDim.x + threadIdx.x;   // BB*EXTV
    if (idx >= (size_t)BB * EXTV) return;
    int b = (int)(idx / EXTV);
    int v = (int)(idx % EXTV);
    float val = 0.0f;
    if (v < VV)
        val = pgen[b] * __expf(logits[(size_t)b * VV + v] - rowmax[b]) / rowsum[b];
    out[((size_t)b * TT + t) * EXTV + v] = val;
}

// scatter copy-distribution: out[b,t, enc_ext[b,s]] += (1-pgen)*attn[b,s]
__global__ void k_scatter(const int* __restrict__ enc_ext, const float* __restrict__ attn,
                          const float* __restrict__ pgen, float* __restrict__ out, int t) {
    int idx = blockIdx.x * blockDim.x + threadIdx.x;   // BB*SS = 6400
    if (idx >= BB * SS) return;
    int b = idx / SS;
    int v = enc_ext[idx];
    if (v < 0) v = 0; if (v >= EXTV) v = EXTV - 1;
    atomicAdd(&out[((size_t)b * TT + t) * EXTV + v], (1.0f - pgen[b]) * attn[idx]);
}

// ---------------------------------------------------------------------------
extern "C" void kernel_launch(void* const* d_in, const int* in_sizes, int n_in,
                              void* d_out, int out_size, void* d_ws, size_t ws_size,
                              hipStream_t stream) {
    const float* enc_output = (const float*)d_in[0];   // [B,S,D]
    const float* dec_hidden = (const float*)d_in[1];   // [B,D]
    const float* emb        = (const float*)d_in[2];   // [V,E]
    const float* gru_k      = (const float*)d_in[3];   // [768,1536]
    const float* gru_rk     = (const float*)d_in[4];   // [512,1536]
    const float* gru_b      = (const float*)d_in[5];   // [2,1536]
    const float* Wfc        = (const float*)d_in[6];   // [512,50000]
    const float* bfc        = (const float*)d_in[7];
    const float* W1         = (const float*)d_in[8];   // [512,512]
    const float* b1         = (const float*)d_in[9];
    const float* W2         = (const float*)d_in[10];  // [512,512]
    const float* b2         = (const float*)d_in[11];
    const float* Va         = (const float*)d_in[12];  // [512,1]
    const float* bV         = (const float*)d_in[13];
    const float* wc         = (const float*)d_in[14];
    const float* wsv        = (const float*)d_in[15];
    const float* wx         = (const float*)d_in[16];
    const float* bp         = (const float*)d_in[17];
    const int*   enc_ext    = (const int*)d_in[18];    // [B,S]
    const int*   dec_inp    = (const int*)d_in[19];    // [B,T]
    float* out = (float*)d_out;                        // [B,T,extV] + [B,D]

    // ---- carve workspace --------------------------------------------------
    char* w = (char*)d_ws;
    auto carve = [&](size_t bytes) -> void* {
        void* p = (void*)w;
        w += (bytes + 255) & ~(size_t)255;
        return p;
    };
    const int NT_FC  = VV / 16;                 // 3125
    const int NT_FCP = (NT_FC + 3) & ~3;        // 3128 (padded to 4)
    const int NT_G   = G3D / 16;                // 96
    const int NT_D   = DD / 16;                 // 32
    const int KB_D   = DD / 32;                 // 16
    const int KB_DE  = DE / 32;                 // 24
    const int MT_ENC = BB * SS / 16;            // 400

    uint32_t* WfcP   = (uint32_t*)carve((size_t)NT_FCP * KB_D  * TILE_WORDS * 4);
    uint32_t* gruKP  = (uint32_t*)carve((size_t)NT_G   * KB_DE * TILE_WORDS * 4);
    uint32_t* gruRKP = (uint32_t*)carve((size_t)NT_G   * KB_D  * TILE_WORDS * 4);
    uint32_t* W1P    = (uint32_t*)carve((size_t)NT_D   * KB_D  * TILE_WORDS * 4);
    uint32_t* W2P    = (uint32_t*)carve((size_t)NT_D   * KB_D  * TILE_WORDS * 4);
    uint32_t* encAP  = (uint32_t*)carve((size_t)MT_ENC * KB_D  * TILE_WORDS * 4);
    float* enc_feat  = (float*)carve((size_t)BB * SS * DD * 4);
    float* logits    = (float*)carve((size_t)BB * VV * 4);
    float* xg        = (float*)carve((size_t)BB * G3D * 4);
    float* hg        = (float*)carve((size_t)BB * G3D * 4);
    float* hbuf      = (float*)carve((size_t)BB * DD * 4);
    uint32_t* hAP    = (uint32_t*)carve((size_t)KB_D  * TILE_WORDS * 4);
    uint32_t* ginAP  = (uint32_t*)carve((size_t)KB_DE * TILE_WORDS * 4);
    float* ctxA      = (float*)carve((size_t)BB * DD * 4);
    float* ctxB      = (float*)carve((size_t)BB * DD * 4);
    float* hW2       = (float*)carve((size_t)BB * DD * 4);
    float* attn      = (float*)carve((size_t)BB * SS * 4);
    float* rowmax    = (float*)carve(BB * 4);
    float* rowsum    = (float*)carve(BB * 4);
    float* pgen      = (float*)carve(BB * 4);

    auto gblk = [](size_t n, int bs) { return (unsigned)((n + bs - 1) / bs); };
    auto ggrp = [](int nt) { return (unsigned)(((nt + 3) / 4 + 3) / 4); }; // 4 tiles/wave, 4 waves/block

    // ---- one-time setup: pack weights to bf16 WMMA layout -----------------
    k_pack_b<<<gblk((size_t)NT_FCP * KB_D  * TILE_WORDS, 256), 256, 0, stream>>>(Wfc,    WfcP,   NT_FCP, KB_D,  VV,  VV);
    k_pack_b<<<gblk((size_t)NT_G   * KB_DE * TILE_WORDS, 256), 256, 0, stream>>>(gru_k,  gruKP,  NT_G,   KB_DE, G3D, G3D);
    k_pack_b<<<gblk((size_t)NT_G   * KB_D  * TILE_WORDS, 256), 256, 0, stream>>>(gru_rk, gruRKP, NT_G,   KB_D,  G3D, G3D);
    k_pack_b<<<gblk((size_t)NT_D   * KB_D  * TILE_WORDS, 256), 256, 0, stream>>>(W1,     W1P,    NT_D,   KB_D,  DD,  DD);
    k_pack_b<<<gblk((size_t)NT_D   * KB_D  * TILE_WORDS, 256), 256, 0, stream>>>(W2,     W2P,    NT_D,   KB_D,  DD,  DD);
    // enc_output as [6400,512] A-operand; enc_feat = enc_output@W1 + b1
    k_pack_a<<<gblk((size_t)MT_ENC * KB_D * TILE_WORDS, 256), 256, 0, stream>>>(
        enc_output, encAP, MT_ENC, KB_D, DD);
    k_gemm_bf16x4<<<dim3(ggrp(NT_D), MT_ENC), 128, 0, stream>>>(encAP, W1P, b1, enc_feat, NT_D, NT_D, KB_D, DD);
    // h = dec_hidden; initial ctx = attention(h) (no p_gen)
    k_copy_f32<<<gblk(BB * DD, 256), 256, 0, stream>>>(dec_hidden, hbuf, BB * DD);
    k_pack_a<<<gblk((size_t)KB_D * TILE_WORDS, 256), 256, 0, stream>>>(hbuf, hAP, 1, KB_D, DD);
    k_gemm_bf16x4<<<dim3(ggrp(NT_D), 1), 128, 0, stream>>>(hAP, W2P, b2, hW2, NT_D, NT_D, KB_D, DD);
    k_attention<<<BB, 512, 0, stream>>>(enc_feat, hW2, enc_output, Va, bV, hbuf,
                                        ctxA, emb, dec_inp, 0,
                                        wc, wsv, wx, bp, attn, ctxA, pgen, 0);

    // ---- 100 decode steps (serial dependency) -----------------------------
    for (int t = 0; t < TT; ++t) {
        float* ctx_cur = (t & 1) ? ctxB : ctxA;
        float* ctx_nxt = (t & 1) ? ctxA : ctxB;
        // gin = [ctx_cur, emb[x_t]] packed straight to A layout
        k_pack_gin<<<gblk((size_t)KB_DE * TILE_WORDS, 256), 256, 0, stream>>>(
            ctx_cur, emb, dec_inp, t, ginAP);
        // GRU: xg = gin@gru_k + b[0]; hg = h@gru_rk + b[1]
        k_gemm_bf16x4<<<dim3(ggrp(NT_G), 1), 128, 0, stream>>>(ginAP, gruKP,  gru_b,       xg, NT_G, NT_G, KB_DE, G3D);
        k_gemm_bf16x4<<<dim3(ggrp(NT_G), 1), 128, 0, stream>>>(hAP,   gruRKP, gru_b + G3D, hg, NT_G, NT_G, KB_D,  G3D);
        k_gru_gates_pack<<<gblk((size_t)KB_D * TILE_WORDS, 256), 256, 0, stream>>>(xg, hg, hbuf, hAP);
        // logits = h@Wfc + bfc (dominant GEMM; bf16 Wfc streams from L2)
        k_gemm_bf16x4<<<dim3(ggrp(NT_FCP), 1), 128, 0, stream>>>(hAP, WfcP, bfc, logits, NT_FC, NT_FCP, KB_D, VV);
        k_softmax_stats<<<BB, 1024, 0, stream>>>(logits, rowmax, rowsum);
        // attention with NEW h (+ fused softmax, ctx, p_gen)
        k_gemm_bf16x4<<<dim3(ggrp(NT_D), 1), 128, 0, stream>>>(hAP, W2P, b2, hW2, NT_D, NT_D, KB_D, DD);
        k_attention<<<BB, 512, 0, stream>>>(enc_feat, hW2, enc_output, Va, bV, hbuf,
                                            ctx_cur, emb, dec_inp, t,
                                            wc, wsv, wx, bp, attn, ctx_nxt, pgen, 1);
        // final distribution for this step
        k_write_out<<<gblk((size_t)BB * EXTV, 256), 256, 0, stream>>>(logits, rowmax, rowsum, pgen, out, t);
        k_scatter<<<gblk(BB * SS, 256), 256, 0, stream>>>(enc_ext, attn, pgen, out, t);
    }

    // h_final appended after final_dists
    k_copy_f32<<<gblk(BB * DD, 256), 256, 0, stream>>>(hbuf, out + (size_t)BB * TT * EXTV, BB * DD);
}
